// InvariantResidualInteraction_27109833572362
// MI455X (gfx1250) — compile-verified, matrix operand order checked by reading, not run on hardware
//
#include <hip/hip_runtime.h>

#define N_NODES 32768
#define N_EDGES 262144
#define C_DIM   64
#define N_LM    16

typedef __attribute__((ext_vector_type(16))) _Float16 v16h;
typedef __attribute__((ext_vector_type(8)))  float    v8f;

union AFrag { v16h v; _Float16 h[16]; };
union CFrag { v8f  v; float    f[8];  };

// K index held by element e (0..15) of an A/B fragment for lane-half `half`.
__device__ __forceinline__ int klocal(int e, int half) {
    return ((e >> 3) << 4) + (half << 3) + (e & 7);
}

// float4-group index (within a 32-wide K chunk) for A-fragment group g (0..3)
__device__ __forceinline__ int agrp(int g, int half) {
    return ((g >> 1) << 2) + half * 2 + (g & 1);
}

// ---------------------------------------------------------------------------
// Zero the `out` region of d_out (atomically accumulated by the edge kernel).
// ---------------------------------------------------------------------------
__global__ void zero_f4_kernel(float4* __restrict__ p, int n4) {
    int i = blockIdx.x * blockDim.x + threadIdx.x;
    if (i < n4) p[i] = make_float4(0.f, 0.f, 0.f, 0.f);
}

// ---------------------------------------------------------------------------
// Pre-permute W_up / W_skip into f16 *fragment order* in workspace:
// element e of lane ln of fragment f=(kc*4+nt) at flat index ((f*32+ln)*16+e),
// so a B operand is one aligned 32-byte load per lane.
// ---------------------------------------------------------------------------
__global__ void prep_weights_kernel(const float* __restrict__ W_up,    // [64,64]
                                    const float* __restrict__ W_skip,  // [640,64]
                                    _Float16* __restrict__ wup16,      // [8*512]
                                    _Float16* __restrict__ wskip16) {  // [80*512]
    int i = blockIdx.x * blockDim.x + threadIdx.x;
    if (i >= 4096 + 40960) return;
    if (i < 4096) {
        int e = i & 15, ln = (i >> 4) & 31, f = i >> 9;
        int nt = f & 3, kc = f >> 2;
        int k = kc * 32 + klocal(e, ln >> 4);
        wup16[i] = (_Float16)W_up[k * 64 + nt * 16 + (ln & 15)];
    } else {
        int j = i - 4096;
        int e = j & 15, ln = (j >> 4) & 31, f = j >> 9;
        int nt = f & 3, kc = f >> 2;                      // kc = 0..19
        int q = kc * 32 + klocal(e, ln >> 4);
        wskip16[j] = (_Float16)W_skip[q * 64 + nt * 16 + (ln & 15)];
    }
}

// ---------------------------------------------------------------------------
// Node kernel: x = feats @ W_up / 8 (to ws), sc = skip-TP (to d_out tail).
// One wave = 16 nodes; kc-outer / nt-inner with 4 live accumulators.
// ---------------------------------------------------------------------------
__global__ __launch_bounds__(128) void node_up_skip_kernel(
    const float* __restrict__ node_attrs,       // [N,10]
    const float* __restrict__ node_feats,       // [N,64]
    const _Float16* __restrict__ wup16,         // fragment-ordered
    const _Float16* __restrict__ wskip16,       // fragment-ordered
    float* __restrict__ x_out,                  // ws [N,64]
    float* __restrict__ sc_out)                 // [N,64]
{
    const int tid  = threadIdx.x;
    const int wave = tid >> 5, lane = tid & 31;
    const int half = lane >> 4, row = lane & 15;
    const int n0   = blockIdx.x * 64 + wave * 16;
    const float*  nfA = node_feats + (n0 + row) * 64;
    const float4* nf4 = (const float4*)nfA;
    const float*  naA = node_attrs + (n0 + row) * 10;

    // ---- linear_up: K = 64 ----
    {
        CFrag acc[4];
        #pragma unroll
        for (int nt = 0; nt < 4; ++nt)
            #pragma unroll
            for (int r = 0; r < 8; ++r) acc[nt].f[r] = 0.f;
        #pragma unroll
        for (int kc = 0; kc < 2; ++kc) {
            AFrag a;
            #pragma unroll
            for (int g = 0; g < 4; ++g) {
                float4 xx = nf4[kc * 8 + agrp(g, half)];
                a.h[g * 4 + 0] = (_Float16)xx.x;
                a.h[g * 4 + 1] = (_Float16)xx.y;
                a.h[g * 4 + 2] = (_Float16)xx.z;
                a.h[g * 4 + 3] = (_Float16)xx.w;
            }
            #pragma unroll
            for (int nt = 0; nt < 4; ++nt) {
                AFrag b;
                b.v = *(const v16h*)&wup16[((kc * 4 + nt) * 32 + lane) * 16];
                acc[nt].v = __builtin_amdgcn_wmma_f32_16x16x32_f16(
                    false, a.v, false, b.v, (short)0, acc[nt].v, false, false);
            }
        }
        #pragma unroll
        for (int nt = 0; nt < 4; ++nt)
            #pragma unroll
            for (int r = 0; r < 8; ++r)
                x_out[(n0 + r + 8 * half) * 64 + nt * 16 + row] =
                    acc[nt].f[r] * 0.125f;
    }

    // ---- skip_tp: K = 640; A[q] = nf[q/10]*na[q%10] built on the fly ----
    {
        CFrag acc[4];
        #pragma unroll
        for (int nt = 0; nt < 4; ++nt)
            #pragma unroll
            for (int r = 0; r < 8; ++r) acc[nt].f[r] = 0.f;
        for (int kc = 0; kc < 20; ++kc) {
            AFrag a;
            const int qb0 = kc * 32 + half * 8;
            #pragma unroll
            for (int e = 0; e < 8; ++e) {
                int q0 = qb0 + e, q1 = qb0 + 16 + e;
                a.h[e]     = (_Float16)(nfA[q0 / 10] * naA[q0 % 10]);
                a.h[e + 8] = (_Float16)(nfA[q1 / 10] * naA[q1 % 10]);
            }
            #pragma unroll
            for (int nt = 0; nt < 4; ++nt) {
                AFrag b;
                b.v = *(const v16h*)&wskip16[((kc * 4 + nt) * 32 + lane) * 16];
                acc[nt].v = __builtin_amdgcn_wmma_f32_16x16x32_f16(
                    false, a.v, false, b.v, (short)0, acc[nt].v, false, false);
            }
        }
        #pragma unroll
        for (int nt = 0; nt < 4; ++nt)
            #pragma unroll
            for (int r = 0; r < 8; ++r)
                sc_out[(n0 + r + 8 * half) * 64 + nt * 16 + row] =
                    acc[nt].f[r] * 0.0395284708f;        // 1/sqrt(640)
    }
}

// ---------------------------------------------------------------------------
// Edge kernel (fused TP + per-l linear + scatter):
//   T[e,l,k] = sum_c (x[s(e),c]*R[e,l,c]) * W_linear[l,c,k]       (WMMA)
//   out[r(e),lm,k] += Y[e,lm] * T[e,l(lm),k] / 128                (atomics)
// ---------------------------------------------------------------------------
__global__ __launch_bounds__(128) void edge_tp_kernel(
    const float* __restrict__ edge_attrs,   // [E,16]  Y
    const float* __restrict__ edge_feats,   // [E,4,64] R
    const int*   __restrict__ sender,       // [E]
    const int*   __restrict__ receiver,     // [E]
    const float* __restrict__ x,            // ws [N,64]
    const float* __restrict__ W_linear,     // [4,64,64]
    float* __restrict__ out)                // [N,16,64]
{
    __shared__ alignas(32) _Float16 sW[32 * 512];   // 32 KB fragment-ordered
    const int tid = threadIdx.x;
    for (int i = tid; i < 32 * 512; i += 128) {
        int e = i & 15, ln = (i >> 4) & 31, f = i >> 9;
        int nt = f & 3, kc = (f >> 2) & 1, l = f >> 3;
        int k = kc * 32 + klocal(e, ln >> 4);
        sW[i] = (_Float16)W_linear[(l * 64 + k) * 64 + nt * 16 + (ln & 15)];
    }
    __syncthreads();

    const int wave = tid >> 5, lane = tid & 31;
    const int half = lane >> 4, row = lane & 15;
    const int e0   = blockIdx.x * 64 + wave * 16;
    const int eA   = e0 + row;

    const float4* x4 = (const float4*)(x + (size_t)sender[eA] * 64);
    const float4* r4 = (const float4*)(edge_feats + (size_t)eA * 256);

    int recv[8];
    #pragma unroll
    for (int r = 0; r < 8; ++r) recv[r] = receiver[e0 + r + 8 * half];

    for (int l = 0; l < 4; ++l) {
        CFrag acc[4];
        #pragma unroll
        for (int nt = 0; nt < 4; ++nt)
            #pragma unroll
            for (int r = 0; r < 8; ++r) acc[nt].f[r] = 0.f;

        #pragma unroll
        for (int kc = 0; kc < 2; ++kc) {
            AFrag a;
            #pragma unroll
            for (int g = 0; g < 4; ++g) {
                int i4 = kc * 8 + agrp(g, half);
                float4 xx = x4[i4];
                float4 rr = r4[l * 16 + i4];
                a.h[g * 4 + 0] = (_Float16)(xx.x * rr.x);
                a.h[g * 4 + 1] = (_Float16)(xx.y * rr.y);
                a.h[g * 4 + 2] = (_Float16)(xx.z * rr.z);
                a.h[g * 4 + 3] = (_Float16)(xx.w * rr.w);
            }
            #pragma unroll
            for (int nt = 0; nt < 4; ++nt) {
                AFrag b;
                b.v = *(const v16h*)&sW[(((l * 2 + kc) * 4 + nt) * 32 + lane) * 16];
                acc[nt].v = __builtin_amdgcn_wmma_f32_16x16x32_f16(
                    false, a.v, false, b.v, (short)0, acc[nt].v, false, false);
            }
        }

        const int lmStart = l * l;
        const int lmCount = 2 * l + 1;
        #pragma unroll
        for (int r = 0; r < 8; ++r) {
            const int er = e0 + r + 8 * half;
            const float t0 = acc[0].f[r] * 0.0078125f;   // 1/(8*16)
            const float t1 = acc[1].f[r] * 0.0078125f;
            const float t2 = acc[2].f[r] * 0.0078125f;
            const float t3 = acc[3].f[r] * 0.0078125f;
            const float* Y = edge_attrs + (size_t)er * 16 + lmStart;
            float* dst = out + ((size_t)recv[r] * 16 + lmStart) * 64 + row;
            for (int j = 0; j < lmCount; ++j) {
                float  y = Y[j];
                float* d = dst + (size_t)j * 64;
                atomicAdd(d,      t0 * y);
                atomicAdd(d + 16, t1 * y);
                atomicAdd(d + 32, t2 * y);
                atomicAdd(d + 48, t3 * y);
            }
        }
    }
}

// ---------------------------------------------------------------------------
extern "C" void kernel_launch(void* const* d_in, const int* in_sizes, int n_in,
                              void* d_out, int out_size, void* d_ws, size_t ws_size,
                              hipStream_t stream) {
    const float* node_attrs = (const float*)d_in[0];
    const float* node_feats = (const float*)d_in[1];
    const float* edge_attrs = (const float*)d_in[2];
    const float* edge_feats = (const float*)d_in[3];
    const int*   edge_index = (const int*)  d_in[4];
    const float* W_up       = (const float*)d_in[5];
    const float* W_linear   = (const float*)d_in[6];
    const float* W_skip     = (const float*)d_in[7];

    float* out  = (float*)d_out;                                  // [N,16,64]
    float* sc   = out + (size_t)N_NODES * N_LM * C_DIM;           // [N,64]

    float*    x_ws    = (float*)d_ws;                             // 8 MB
    _Float16* wup16   = (_Float16*)((char*)d_ws +
                         (size_t)N_NODES * C_DIM * sizeof(float));
    _Float16* wskip16 = wup16 + 8 * 512;

    // 1) zero the atomically-accumulated `out` region
    const int n4 = (N_NODES * N_LM * C_DIM) / 4;
    zero_f4_kernel<<<(n4 + 255) / 256, 256, 0, stream>>>((float4*)out, n4);

    // 2) pre-permute weights into fragment-ordered f16
    prep_weights_kernel<<<(4096 + 40960 + 255) / 256, 256, 0, stream>>>(
        W_up, W_skip, wup16, wskip16);

    // 3) node-level GEMMs (x -> ws, sc -> d_out tail)
    node_up_skip_kernel<<<N_NODES / 64, 128, 0, stream>>>(
        node_attrs, node_feats, wup16, wskip16, x_ws, sc);

    // 4) fused edge tensor-product + linear + scatter
    edge_tp_kernel<<<N_EDGES / 64, 128, 0, stream>>>(
        edge_attrs, edge_feats, edge_index, edge_index + N_EDGES,
        x_ws, W_linear, out);
}
